// GraphTransformerLayer_31404800868534
// MI455X (gfx1250) — matrix operand order repaired and checked by reading, not compiled, not run on hardware
//
#include <hip/hip_runtime.h>
#include <hip/hip_bf16.h>

typedef __attribute__((ext_vector_type(16))) __bf16        v16bf;
typedef __attribute__((ext_vector_type(8)))  float         v8f;
typedef __attribute__((ext_vector_type(8)))  unsigned int  v8u;
typedef __attribute__((ext_vector_type(4)))  int           v4i;

#if defined(__gfx1250__) && __has_builtin(__builtin_amdgcn_global_load_async_to_lds_b128)
#define HAVE_ASYNC_LDS 1
typedef __attribute__((address_space(1))) v4i* gv4_t;   // global 128-bit vector ptr
typedef __attribute__((address_space(3))) v4i* lv4_t;   // LDS    128-bit vector ptr
#endif

// ---------- helpers ----------
__device__ __forceinline__ unsigned short f2bf(float x) {
    unsigned int u = __builtin_bit_cast(unsigned int, x);
    u += 0x7FFFu + ((u >> 16) & 1u);          // round-to-nearest-even bf16
    return (unsigned short)(u >> 16);
}
__device__ __forceinline__ unsigned int packbf(float lo, float hi) {
    return (unsigned int)f2bf(lo) | ((unsigned int)f2bf(hi) << 16);
}
__device__ __forceinline__ v16bf as_bf16x16(v8u u) {
    return __builtin_bit_cast(v16bf, u);
}
__device__ __forceinline__ v8f wmma_bf16(v16bf a, v16bf b, v8f c) {
    // D = A(16x32 bf16) * B(32x16 bf16) + C(16x16 f32)
    return __builtin_amdgcn_wmma_f32_16x16x32_bf16(false, a, false, b, (short)0, c, false, false);
}

// ---------------------------------------------------------------------------
// Kernel 1: repack Wq|Wk|Wv (each [64,256]) and Wo ([256,64]) into per-lane
// WMMA B-fragment order: fragment = 256 dwords, dword index = lane*8 + v,
// value packs bf16(W[k][n]), bf16(W[k+1][n]) with
//   n = coltile*16 + (lane&15),  k = kfrag*32 + (v<4 ? 0 : 16) + (lane<16 ? 0 : 8) + 2(v&3)
// Fragments 0..95  : QKV, index = (mat*16 + coltile)*2 + kfrag
// Fragments 96..127: Wo,  index = 96 + coltile*8 + kfrag
// ---------------------------------------------------------------------------
__global__ __launch_bounds__(256) void gt_prepack(const float* __restrict__ Wq,
                                                  const float* __restrict__ Wk,
                                                  const float* __restrict__ Wv,
                                                  const float* __restrict__ Wo,
                                                  unsigned int* __restrict__ wpack) {
    int gid  = blockIdx.x * 256 + threadIdx.x;   // 0 .. 32767
    int frag = gid >> 8;
    int lane = (gid >> 3) & 31;
    int v    = gid & 7;
    unsigned int val;
    if (frag < 96) {
        int m  = frag >> 5;            // 0=Wq 1=Wk 2=Wv
        int c  = (frag >> 1) & 15;     // col tile within 256 cols
        int f  = frag & 1;             // k fragment (K=0..31 / 32..63)
        int nn = c * 16 + (lane & 15);
        int k  = f * 32 + ((v < 4) ? 0 : 16) + ((lane < 16) ? 0 : 8) + 2 * (v & 3);
        const float* W = (m == 0) ? Wq : (m == 1) ? Wk : Wv;
        val = packbf(W[k * 256 + nn], W[(k + 1) * 256 + nn]);
    } else {
        int fo = frag - 96;
        int t  = fo >> 3;              // col tile within 64 cols
        int kf = fo & 7;               // k fragment (K = kf*32 ..)
        int nn = t * 16 + (lane & 15);
        int k  = kf * 32 + ((v < 4) ? 0 : 16) + ((lane < 16) ? 0 : 8) + 2 * (v & 3);
        val = packbf(Wo[k * 64 + nn], Wo[(k + 1) * 64 + nn]);
    }
    wpack[gid] = val;
}

// ---------------------------------------------------------------------------
// Kernel 2: zero h_new accumulator
// ---------------------------------------------------------------------------
__global__ __launch_bounds__(256) void gt_zero(float4* __restrict__ p, int n4) {
    int i = blockIdx.x * 256 + threadIdx.x;
    if (i < n4) p[i] = make_float4(0.f, 0.f, 0.f, 0.f);
}

// ---------------------------------------------------------------------------
// Kernel 3: QKV = h @ [Wq|Wk|Wv].  8 waves/block, one 16-row tile per wave,
// 48 col-tiles (Q:0-15, K:16-31, V:32-47), 2 bf16 WMMAs per tile (K=64).
// h rows staged through LDS with async-to-LDS when available.
// ---------------------------------------------------------------------------
__global__ __launch_bounds__(256) void gt_qkv(const float* __restrict__ h,
                                              const unsigned int* __restrict__ wpack,
                                              float* __restrict__ Q,
                                              float* __restrict__ K,
                                              float* __restrict__ V,
                                              int n) {
    __shared__ float hs[128 * 64];                       // 32 KB
    int rb_block = blockIdx.x * 128;

#ifdef HAVE_ASYNC_LDS
    {
        // 2048 x b128 transfers, 256 threads x 8 iters
        for (int it = 0; it < 8; ++it) {
            int idx = it * 256 + threadIdx.x;            // b128 index
            int r   = rb_block + (idx >> 4);             // 16 x b128 per row
            if (r > n - 1) r = n - 1;
            const float* gsrc = h + (size_t)r * 64 + (idx & 15) * 4;
            __builtin_amdgcn_global_load_async_to_lds_b128(
                (gv4_t)gsrc, (lv4_t)&hs[idx * 4], 0, 0);
        }
#if __has_builtin(__builtin_amdgcn_s_wait_asynccnt)
        __builtin_amdgcn_s_wait_asynccnt(0);
#else
        asm volatile("s_wait_asynccnt 0x0" ::: "memory");
#endif
    }
#else
    for (int i = threadIdx.x; i < 128 * 64; i += 256) {  // coalesced row stage
        int r = rb_block + (i >> 6);
        if (r > n - 1) r = n - 1;
        hs[i] = h[(size_t)r * 64 + (i & 63)];
    }
#endif
    __syncthreads();

    int wave = threadIdx.x >> 5;
    int lane = threadIdx.x & 31;
    int rb   = wave * 16;            // row base inside LDS tile
    int grb  = rb_block + rb;        // global row base
    int mrow = lane & 15;            // A: lanes 0-15 and 16-31 both map M=0..15
    int koff = (lane < 16) ? 0 : 8;  // A: high lanes hold K+8 group

    // Build the two 16x32 bf16 A fragments (K = 0..31 and 32..63)
    v16bf afrag[2];
    for (int f = 0; f < 2; ++f) {
        v8u au;
        for (int v = 0; v < 8; ++v) {
            int k = f * 32 + ((v < 4) ? 0 : 16) + koff + 2 * (v & 3);
            const float* hp = &hs[(rb + mrow) * 64 + k];
            au[v] = packbf(hp[0], hp[1]);
        }
        afrag[f] = as_bf16x16(au);
    }

    int  m0   = (lane < 16) ? 0 : 8;
    int  nc   = lane & 15;
    bool full = (grb + 16 <= n);     // wave-uniform: all 16 rows in range
    for (int c = 0; c < 48; ++c) {
        const v8u* bp = reinterpret_cast<const v8u*>(wpack + (size_t)c * 2 * 256 + lane * 8);
        v8f acc = {};
        acc = wmma_bf16(afrag[0], as_bf16x16(bp[0]),  acc);
        acc = wmma_bf16(afrag[1], as_bf16x16(bp[32]), acc);   // +256 dwords
        float* outp = (c < 16) ? Q : (c < 32) ? K : V;
        int cb = (c & 15) * 16;
        float* op0 = outp + (size_t)(grb + m0) * 256 + cb + nc;
        if (full) {
#pragma unroll
            for (int r = 0; r < 8; ++r) op0[(size_t)r * 256] = acc[r];
        } else {
            for (int r = 0; r < 8; ++r)
                if (grb + m0 + r < n) op0[(size_t)r * 256] = acc[r];
        }
    }
}

// ---------------------------------------------------------------------------
// Kernel 4: edge phase. One wave32 per edge, 4 edges per wave with
// next-edge prefetch (global_prefetch_b8). Lane l owns dims [8l, 8l+8)
// of the 256-wide (H=4, D=64) feature vector -> lanes 8h..8h+7 form head h.
// ---------------------------------------------------------------------------
#define EDGES_PER_WAVE 4
__global__ __launch_bounds__(256) void gt_edge(const int* __restrict__ src,
                                               const int* __restrict__ dst,
                                               const float* __restrict__ Q,
                                               const float* __restrict__ K,
                                               const float* __restrict__ V,
                                               float* __restrict__ hnew,
                                               int e_total) {
    int wave = threadIdx.x >> 5;
    int lane = threadIdx.x & 31;
    long base = ((long)blockIdx.x * 8 + wave) * EDGES_PER_WAVE;
    if (base >= e_total) return;                    // wave-uniform

    for (int i = 0; i < EDGES_PER_WAVE; ++i) {
        long e = base + i;
        if (e >= e_total) break;                    // wave-uniform
        int cs = src[e], cd = dst[e];
        // prefetch next edge's gather rows while working on this one
        if (i + 1 < EDGES_PER_WAVE && e + 1 < e_total) {
            int s2 = src[e + 1], d2 = dst[e + 1];
            __builtin_prefetch(Q + (size_t)s2 * 256 + lane * 8, 0, 0);
            __builtin_prefetch(K + (size_t)d2 * 256 + lane * 8, 0, 0);
            __builtin_prefetch(V + (size_t)s2 * 256 + lane * 8, 0, 0);
        }

        const float4* q4 = reinterpret_cast<const float4*>(Q + (size_t)cs * 256 + lane * 8);
        const float4* k4 = reinterpret_cast<const float4*>(K + (size_t)cd * 256 + lane * 8);
        float4 qa = q4[0], qb = q4[1];
        float4 ka = k4[0], kb = k4[1];
        float part = qa.x * ka.x + qa.y * ka.y + qa.z * ka.z + qa.w * ka.w +
                     qb.x * kb.x + qb.y * kb.y + qb.z * kb.z + qb.w * kb.w;
        // reduce within the 8 lanes of this head
        part += __shfl_xor(part, 1, 32);
        part += __shfl_xor(part, 2, 32);
        part += __shfl_xor(part, 4, 32);
        float score = part * 0.125f;                // 1/sqrt(64)
        // softmax over the 4 heads (cross-group xor 8, 16)
        float mx = fmaxf(score, __shfl_xor(score, 8, 32));
        mx = fmaxf(mx, __shfl_xor(mx, 16, 32));
        float ex = __expf(score - mx);
        float sm = ex + __shfl_xor(ex, 8, 32);
        sm += __shfl_xor(sm, 16, 32);
        float attn = ex / sm;

        const float4* v4 = reinterpret_cast<const float4*>(V + (size_t)cs * 256 + lane * 8);
        float4 va = v4[0], vb = v4[1];
        float* op = hnew + (size_t)cd * 256 + lane * 8;
        unsafeAtomicAdd(op + 0, va.x * attn);
        unsafeAtomicAdd(op + 1, va.y * attn);
        unsafeAtomicAdd(op + 2, va.z * attn);
        unsafeAtomicAdd(op + 3, va.w * attn);
        unsafeAtomicAdd(op + 4, vb.x * attn);
        unsafeAtomicAdd(op + 5, vb.y * attn);
        unsafeAtomicAdd(op + 6, vb.z * attn);
        unsafeAtomicAdd(op + 7, vb.w * attn);
    }
}

// ---------------------------------------------------------------------------
// Kernel 5: h_proj = h_new @ Wo, + residual, LayerNorm.
// One 16-row tile per wave: 8 K-fragments x 4 col-tiles of bf16 WMMA,
// then residual+LN via LDS.
// ---------------------------------------------------------------------------
__global__ __launch_bounds__(256) void gt_out(const float* __restrict__ hnew,
                                              const unsigned int* __restrict__ wpack_o,
                                              const float* __restrict__ h,
                                              const float* __restrict__ gamma,
                                              const float* __restrict__ beta,
                                              float* __restrict__ out,
                                              int n) {
    __shared__ float xs[128 * 64];                  // 32 KB
    int wave = threadIdx.x >> 5;
    int lane = threadIdx.x & 31;
    int grb  = blockIdx.x * 128 + wave * 16;
    int mrow = lane & 15;
    int koff = (lane < 16) ? 0 : 8;
    int arow = grb + mrow; if (arow > n - 1) arow = n - 1;
    const float* ap = hnew + (size_t)arow * 256;

    v8f acc[4] = {v8f{}, v8f{}, v8f{}, v8f{}};
    for (int kf = 0; kf < 8; ++kf) {
        v8u au;
        for (int v = 0; v < 8; ++v) {
            int k = kf * 32 + ((v < 4) ? 0 : 16) + koff + 2 * (v & 3);
            au[v] = packbf(ap[k], ap[k + 1]);
        }
        v16bf a = as_bf16x16(au);
        for (int t = 0; t < 4; ++t) {
            const v8u* bp = reinterpret_cast<const v8u*>(wpack_o + (size_t)(t * 8 + kf) * 256 + lane * 8);
            acc[t] = wmma_bf16(a, as_bf16x16(bp[0]), acc[t]);
        }
    }

    // residual add -> LDS (wave-uniform fast path; guarded only in tail wave)
    int  m0   = (lane < 16) ? 0 : 8;
    int  nc   = lane & 15;
    bool full = (grb + 16 <= n);
    if (full) {
#pragma unroll
        for (int t = 0; t < 4; ++t) {
#pragma unroll
            for (int r = 0; r < 8; ++r) {
                int lm = wave * 16 + m0 + r;
                int grow = blockIdx.x * 128 + lm;
                xs[lm * 64 + t * 16 + nc] = acc[t][r] + h[(size_t)grow * 64 + t * 16 + nc];
            }
        }
    } else {
        for (int t = 0; t < 4; ++t) {
            for (int r = 0; r < 8; ++r) {
                int lm = wave * 16 + m0 + r;
                int grow = blockIdx.x * 128 + lm;
                float resid = (grow < n) ? h[(size_t)grow * 64 + t * 16 + nc] : 0.f;
                xs[lm * 64 + t * 16 + nc] = acc[t][r] + resid;
            }
        }
    }
    __syncthreads();

    // LayerNorm: one row per thread (threads 0..127)
    if (threadIdx.x < 128) {
        int lm   = threadIdx.x;
        int grow = blockIdx.x * 128 + lm;
        if (grow < n) {
            const float* xr = &xs[lm * 64];
            float mu = 0.f;
            for (int d2 = 0; d2 < 64; ++d2) mu += xr[d2];
            mu *= (1.f / 64.f);
            float var = 0.f;
            for (int d2 = 0; d2 < 64; ++d2) { float t2 = xr[d2] - mu; var += t2 * t2; }
            var *= (1.f / 64.f);
            float rs = rsqrtf(var + 1e-5f);
            float* orow = out + (size_t)grow * 64;
            for (int d2 = 0; d2 < 64; ++d2)
                orow[d2] = (xr[d2] - mu) * rs * gamma[d2] + beta[d2];
        }
    }
}

// ---------------------------------------------------------------------------
extern "C" void kernel_launch(void* const* d_in, const int* in_sizes, int n_in,
                              void* d_out, int out_size, void* d_ws, size_t ws_size,
                              hipStream_t stream) {
    const float* h     = (const float*)d_in[0];
    const int*   src   = (const int*)  d_in[1];
    const int*   dst   = (const int*)  d_in[2];
    const float* Wq    = (const float*)d_in[3];
    const float* Wk    = (const float*)d_in[4];
    const float* Wv    = (const float*)d_in[5];
    const float* Wo    = (const float*)d_in[6];
    const float* gamma = (const float*)d_in[7];
    const float* beta  = (const float*)d_in[8];

    int n = in_sizes[0] / 64;      // 50000
    int e = in_sizes[1];           // 800000

    size_t per = (size_t)n * 256;  // floats per [N, H*D] buffer
    float* Q    = (float*)d_ws;
    float* K    = Q + per;
    float* V    = K + per;
    float* hnew = V + per;
    unsigned int* wpack = (unsigned int*)(hnew + per);   // 128 fragments * 1KB

    gt_prepack<<<128, 256, 0, stream>>>(Wq, Wk, Wv, Wo, wpack);

    int n4 = (int)(per / 4);
    gt_zero<<<(n4 + 255) / 256, 256, 0, stream>>>((float4*)hnew, n4);

    int nblk = (n + 127) / 128;
    gt_qkv<<<nblk, 256, 0, stream>>>(h, wpack, Q, K, V, n);

    int eblk = (e + 8 * EDGES_PER_WAVE - 1) / (8 * EDGES_PER_WAVE);
    gt_edge<<<eblk, 256, 0, stream>>>(src, dst, Q, K, V, hnew, e);

    gt_out<<<nblk, 256, 0, stream>>>(hnew, wpack + 96 * 256, h, gamma, beta,
                                     (float*)d_out, n);
}
